// BinaryLayer_48060684042318
// MI455X (gfx1250) — 8-bit matrix operand order repaired and checked by reading, compiled, not run on hardware
//
#include <hip/hip_runtime.h>
#include <hip/hip_bf16.h>

#define B_ROWS   4096
#define F_DIM    784
#define OUT_N    128
#define OR_T     32
#define AND_T    16
#define N_TERMS  (OUT_N * OR_T)
#define K_PAD    832
#define KT       13
#define BPAIRS   (B_ROWS / 32)
#define TGROUPS  (N_TERMS / 64)
#define PACK_BYTES (3407872)

typedef int v8i __attribute__((ext_vector_type(8)));

__global__ void bl_zero_kernel(int4* __restrict__ p, int n4) {
    int i = blockIdx.x * blockDim.x + threadIdx.x;
    if (i < n4) p[i] = make_int4(0, 0, 0, 0);
}

__global__ void bl_build_rows_kernel(const int* __restrict__ w,
                                     signed char* __restrict__ Mtmp,
                                     int* __restrict__ target) {
    int t = blockIdx.x * blockDim.x + threadIdx.x;
    if (t >= N_TERMS) return;
    signed char* row = Mtmp + (size_t)t * K_PAD;
    int c = 0;
    bool allzero = true;
    #pragma unroll
    for (int s = 0; s < AND_T; ++s) {
        int idx = w[t * AND_T + s];
        if (idx != 0) allzero = false;
        if (idx == 0) {
            c += 1;
        } else if (idx <= F_DIM) {
            row[idx - 1] += 1;
        } else {
            c += 1;
            row[idx - F_DIM - 1] -= 1;
        }
    }
    target[t] = allzero ? 0x7fffffff : (AND_T - c);
}

__global__ void bl_pack_b_kernel(const signed char* __restrict__ M,
                                 signed char* __restrict__ Bp, int total) {
    int i = blockIdx.x * blockDim.x + threadIdx.x;
    if (i >= total) return;
    int byteInLane = i & 31;
    int v = byteInLane >> 2, j = byteInLane & 3;
    int lane = (i >> 5) & 31;
    int tp = i >> 10;
    int k_tile = tp % KT;
    int t_tile = tp / KT;
    int col = lane & 15, khalf = lane >> 4;
    int g = v >> 2, ii = v & 3;
    int k = k_tile * 64 + g * 32 + khalf * 16 + ii * 4 + j;
    int term = t_tile * 16 + col;
    Bp[i] = M[(size_t)term * K_PAD + k];
}

__global__ void bl_pack_a_kernel(const int* __restrict__ x,
                                 unsigned char* __restrict__ Ap, int total) {
    int i = blockIdx.x * blockDim.x + threadIdx.x;
    if (i >= total) return;
    int byteInLane = i & 31;
    int lane = (i >> 5) & 31;
    int tp = i >> 10;
    int k_tile = tp % KT;
    int b_tile = tp / KT;
    int m = b_tile * 16 + (lane & 15);
    int kbase = (lane >> 4) * 16;
    int p = byteInLane >> 4, off = byteInLane & 15;
    int k = k_tile * 64 + p * 32 + kbase + off;
    unsigned char val = 0;
    if (k < F_DIM) val = (unsigned char)x[(size_t)m * F_DIM + k];
    Ap[i] = val;
}

__global__ __launch_bounds__(256) void bl_gemm_kernel(
        const unsigned char* __restrict__ Ap,
        const signed char* __restrict__ Bp,
        const int* __restrict__ target,
        float* __restrict__ out) {
    const int lane = threadIdx.x & 31;
    const int wave = threadIdx.x >> 5;
    const int w = blockIdx.x * 8 + wave;
    const int b_pair  = w & (BPAIRS - 1);
    const int t_group = w >> 7;
    const int t_tile0 = t_group * 4;
    const int b_tile0 = b_pair * 2;

    v8i acc00 = {}, acc01 = {}, acc02 = {}, acc03 = {};
    v8i acc10 = {}, acc11 = {}, acc12 = {}, acc13 = {};

    const size_t laneOff = (size_t)lane * 32;
    const unsigned char* aPtr0 = Ap + ((size_t)(b_tile0 + 0) * KT) * 1024 + laneOff;
    const unsigned char* aPtr1 = Ap + ((size_t)(b_tile0 + 1) * KT) * 1024 + laneOff;
    const signed char*   bPtr0 = Bp + ((size_t)(t_tile0 + 0) * KT) * 1024 + laneOff;
    const signed char*   bPtr1 = Bp + ((size_t)(t_tile0 + 1) * KT) * 1024 + laneOff;
    const signed char*   bPtr2 = Bp + ((size_t)(t_tile0 + 2) * KT) * 1024 + laneOff;
    const signed char*   bPtr3 = Bp + ((size_t)(t_tile0 + 3) * KT) * 1024 + laneOff;

    #pragma unroll
    for (int kt = 0; kt < KT; ++kt) {
        const size_t o = (size_t)kt * 1024;
        v8i a0 = *(const v8i*)(aPtr0 + o);
        v8i a1 = *(const v8i*)(aPtr1 + o);
        v8i b0 = *(const v8i*)(bPtr0 + o);
        v8i b1 = *(const v8i*)(bPtr1 + o);
        v8i b2 = *(const v8i*)(bPtr2 + o);
        v8i b3 = *(const v8i*)(bPtr3 + o);
        acc00 = __builtin_amdgcn_wmma_i32_16x16x64_iu8(false, a0, true, b0, acc00, false, false);
        acc01 = __builtin_amdgcn_wmma_i32_16x16x64_iu8(false, a0, true, b1, acc01, false, false);
        acc02 = __builtin_amdgcn_wmma_i32_16x16x64_iu8(false, a0, true, b2, acc02, false, false);
        acc03 = __builtin_amdgcn_wmma_i32_16x16x64_iu8(false, a0, true, b3, acc03, false, false);
        acc10 = __builtin_amdgcn_wmma_i32_16x16x64_iu8(false, a1, true, b0, acc10, false, false);
        acc11 = __builtin_amdgcn_wmma_i32_16x16x64_iu8(false, a1, true, b1, acc11, false, false);
        acc12 = __builtin_amdgcn_wmma_i32_16x16x64_iu8(false, a1, true, b2, acc12, false, false);
        acc13 = __builtin_amdgcn_wmma_i32_16x16x64_iu8(false, a1, true, b3, acc13, false, false);
    }

    const int col = lane & 15;
    const int tg0 = target[(t_tile0 + 0) * 16 + col];
    const int tg1 = target[(t_tile0 + 1) * 16 + col];
    const int tg2 = target[(t_tile0 + 2) * 16 + col];
    const int tg3 = target[(t_tile0 + 3) * 16 + col];

    const int row  = lane & 15;
    const int half = lane >> 4;
    const int outIdx = t_group * 2 + half;
    const size_t outBase = (size_t)(b_tile0 * 16 + row) * OUT_N + outIdx;

#define BL_EPILOGUE(I)                                                          \
    do {                                                                        \
        unsigned fA = 0, fB = 0;         \
        _Pragma("unroll")                                                       \
        for (int r = 0; r < 8; ++r) {                                           \
            unsigned m0 = __builtin_amdgcn_ballot_w32(acc##I##0[r] == tg0);     \
            unsigned m1 = __builtin_amdgcn_ballot_w32(acc##I##1[r] == tg1);     \
            unsigned m2 = __builtin_amdgcn_ballot_w32(acc##I##2[r] == tg2);     \
            unsigned m3 = __builtin_amdgcn_ballot_w32(acc##I##3[r] == tg3);     \
            unsigned mA = m0 | m1, mB = m2 | m3;                                \
            fA |= ((mA & 0xffffu) ? 1u : 0u) << r;                              \
            fA |= ((mA >> 16)     ? 1u : 0u) << (r + 8);                        \
            fB |= ((mB & 0xffffu) ? 1u : 0u) << r;                              \
            fB |= ((mB >> 16)     ? 1u : 0u) << (r + 8);                        \
        }                                                                       \
        const unsigned flags = half ? fB : fA;                                  \
        out[outBase + (size_t)(I) * 16 * OUT_N] = (float)((flags >> row) & 1u); \
    } while (0)

    BL_EPILOGUE(0);
    BL_EPILOGUE(1);
#undef BL_EPILOGUE
}

extern "C" void kernel_launch(void* const* d_in, const int* in_sizes, int n_in,
                              void* d_out, int out_size, void* d_ws, size_t ws_size,
                              hipStream_t stream) {
    const int* x = (const int*)d_in[0];
    const int* w = (const int*)d_in[1];
    float* out = (float*)d_out;

    char* ws = (char*)d_ws;
    signed char*   Mtmp   = (signed char*)(ws);
    signed char*   Bp     = (signed char*)(ws + 1 * PACK_BYTES);
    unsigned char* Ap     = (unsigned char*)(ws + 2 * PACK_BYTES);
    int*           target = (int*)(ws + 3 * (size_t)PACK_BYTES);

    {
        int n4 = PACK_BYTES / 16;
        bl_zero_kernel<<<(n4 + 255) / 256, 256, 0, stream>>>((int4*)Mtmp, n4);
    }
    bl_build_rows_kernel<<<N_TERMS / 256, 256, 0, stream>>>(w, Mtmp, target);
    bl_pack_b_kernel<<<(PACK_BYTES + 255) / 256, 256, 0, stream>>>(Mtmp, Bp, PACK_BYTES);
    bl_pack_a_kernel<<<(PACK_BYTES + 255) / 256, 256, 0, stream>>>(x, Ap, PACK_BYTES);
    bl_gemm_kernel<<<(BPAIRS * TGROUPS) / 8, 256, 0, stream>>>(Ap, Bp, target, out);
}
